// RTFSBlock_27693949124682
// MI455X (gfx1250) — matrix operand order repaired and checked by reading, not compiled, not run on hardware
//
#include <hip/hip_runtime.h>
#include <math.h>

// ---------------------------------------------------------------------------
// Problem constants (from the reference)
// ---------------------------------------------------------------------------
#define BB    8
#define CC    256
#define TT    256
#define FFQ   128
#define DD    64
#define HH    128
#define NHEAD 4
#define KK    8
#define II    512            // D*K
#define H3    384            // 3*H

// GEMM tiling: 8 waves (wave32) -> 128x64 tile, K-chunk 32 (one bf16 WMMA)
#define TM 128
#define TN 64
#define TK 32

typedef __bf16 bf16_t;
typedef bf16_t v16bf __attribute__((ext_vector_type(16)));
typedef float  v8f   __attribute__((ext_vector_type(8)));

union FragU { v16bf v; unsigned short u[16]; };

__device__ __forceinline__ unsigned short f2bf(float f) {
  union { float f; unsigned int u; } x; x.f = f;
  unsigned int r = (x.u + 0x7FFFu + ((x.u >> 16) & 1u)) >> 16;  // RNE
  return (unsigned short)r;
}

__device__ __forceinline__ unsigned int pack2bf(float a, float b) {
  return (unsigned int)f2bf(a) | ((unsigned int)f2bf(b) << 16);
}

// A-fragment (16x32, 16-bit): lane&15 = M row.
// u[g*8 + j] = elem(M = lane&15, K = (lane>=16 ? 16 : 0) + g*8 + j)
__device__ __forceinline__ v16bf frag_ld_a(const unsigned short* tile, int base, int lane) {
  FragU f;
  const unsigned short* row = tile + (size_t)(base + (lane & 15)) * TK;
  const int half8 = ((lane >> 4) & 1) * 8;
#pragma unroll
  for (int g = 0; g < 2; ++g)
#pragma unroll
    for (int j = 0; j < 8; ++j)
      f.u[g * 8 + j] = row[half8 * 2 + g * 8 + j];
  return f.v;
}

// B-fragment (32x16, 16-bit): lane&15 = N col; lanes 0-15 K=0..15, 16-31 K=16..31.
// Tile stored transposed as (N, K).
__device__ __forceinline__ v16bf frag_ld_b(const unsigned short* tile, int base, int lane) {
  FragU f;
  const unsigned short* row =
      tile + (size_t)(base + (lane & 15)) * TK + ((lane >> 4) & 1) * 16;
#pragma unroll
  for (int j = 0; j < 16; ++j) f.u[j] = row[j];
  return f.v;
}

// ---------------------------------------------------------------------------
// Tile staging: fp32 global -> packed bf16 pairs in LDS (one b32 per store).
// Interior tiles run a completely unguarded loop (no per-element branches);
// guarded variants only execute for edge tiles. NCHW gather is division-free
// on the fast path (requires Sa >= TM, true for all uses here).
// ---------------------------------------------------------------------------
template<int ANCHW>
__device__ __forceinline__ void stage_tiles(
    unsigned int* __restrict__ As32, unsigned int* __restrict__ Bs32,
    const float* __restrict__ A, const float* __restrict__ Bw,
    int M, int N, int Kd, int Sa, int m0, int n0, int k0, int tid)
{
  const bool fullK = (k0 + TK) <= Kd;
  const bool fullM = (m0 + TM) <= M;
  const bool fullN = (n0 + TN) <= N;

  // ---- A tile: 128x32 -> 2048 packed pairs, 8 per thread ----
  if (ANCHW) {
    const int bb0 = m0 / Sa;
    const int ss0 = m0 - bb0 * Sa;
    if (fullM && fullK) {
#pragma unroll
      for (int e = 0; e < 8; ++e) {
        int p = tid + e * 256;
        int r = p >> 4, kk = (p & 15) * 2;
        int bb = bb0, ss = ss0 + r;
        if (ss >= Sa) { ss -= Sa; ++bb; }          // Sa >= TM -> single wrap
        const float* base = A + ((size_t)bb * Kd + (k0 + kk)) * Sa + ss;
        As32[p] = pack2bf(base[0], base[Sa]);
      }
    } else {
#pragma unroll
      for (int e = 0; e < 8; ++e) {
        int p = tid + e * 256;
        int r = p >> 4, kk = (p & 15) * 2;
        int gm = m0 + r, gk = k0 + kk;
        float v0 = 0.f, v1 = 0.f;
        if (gm < M) {
          int bb = gm / Sa, ss = gm - bb * Sa;
          const float* base = A + ((size_t)bb * Kd + gk) * Sa + ss;
          if (gk < Kd)     v0 = base[0];
          if (gk + 1 < Kd) v1 = base[Sa];
        }
        As32[p] = pack2bf(v0, v1);
      }
    }
  } else {
    if (fullM && fullK) {
#pragma unroll
      for (int e = 0; e < 8; ++e) {
        int p = tid + e * 256;
        int r = p >> 4, kk = (p & 15) * 2;
        const float* base = A + (size_t)(m0 + r) * Kd + (k0 + kk);
        As32[p] = pack2bf(base[0], base[1]);       // merged global_load_b64
      }
    } else {
#pragma unroll
      for (int e = 0; e < 8; ++e) {
        int p = tid + e * 256;
        int r = p >> 4, kk = (p & 15) * 2;
        int gm = m0 + r, gk = k0 + kk;
        float v0 = 0.f, v1 = 0.f;
        if (gm < M) {
          const float* base = A + (size_t)gm * Kd + gk;
          if (gk < Kd)     v0 = base[0];
          if (gk + 1 < Kd) v1 = base[1];
        }
        As32[p] = pack2bf(v0, v1);
      }
    }
  }

  // ---- B tile: 64x32 -> 1024 packed pairs, 4 per thread ----
  if (fullN && fullK) {
#pragma unroll
    for (int e = 0; e < 4; ++e) {
      int p = tid + e * 256;
      int r = p >> 4, kk = (p & 15) * 2;
      const float* base = Bw + (size_t)(n0 + r) * Kd + (k0 + kk);
      Bs32[p] = pack2bf(base[0], base[1]);
    }
  } else {
#pragma unroll
    for (int e = 0; e < 4; ++e) {
      int p = tid + e * 256;
      int r = p >> 4, kk = (p & 15) * 2;
      int gn = n0 + r, gk = k0 + kk;
      float v0 = 0.f, v1 = 0.f;
      if (gn < N) {
        const float* base = Bw + (size_t)gn * Kd + gk;
        if (gk < Kd)     v0 = base[0];
        if (gk + 1 < Kd) v1 = base[1];
      }
      Bs32[p] = pack2bf(v0, v1);
    }
  }
}

template<int ONCHW>
__device__ __forceinline__ void store_elem(
    float* __restrict__ C, const float* __restrict__ bias,
    const float* __restrict__ resid, int relu, int N, int So,
    int gm, int gn, float v)
{
  if (bias) v += bias[gn];
  if (relu) v = fmaxf(v, 0.f);
  size_t oidx;
  if (ONCHW) {
    int bb = gm / So, ss = gm - bb * So;
    oidx = ((size_t)bb * N + gn) * So + ss;
  } else {
    oidx = (size_t)gm * N + gn;
  }
  if (resid) v += resid[oidx];
  C[oidx] = v;
}

// ---------------------------------------------------------------------------
// Universal GEMM (NT):  C(MxN) = A(MxKd) * B(NxKd)^T  [+bias][+resid][ReLU]
//   ANCHW : A gathered from NCHW (batch = m/Sa, spatial = m%Sa, chan = k)
//   ONCHW : C scattered to NCHW (chan = n, spatial = m%So)
// bf16 LDS staging (double-buffered, one barrier per K-chunk),
// v_wmma_f32_16x16x32_bf16, fp32 accumulate.
// ---------------------------------------------------------------------------
template<int ANCHW, int ONCHW>
__global__ void __launch_bounds__(256)
gemm_nt_wmma(const float* __restrict__ A, const float* __restrict__ Bw,
             float* __restrict__ C, const float* __restrict__ bias,
             const float* __restrict__ resid,
             int M, int N, int Kd, int Sa, int So, int relu)
{
  __shared__ unsigned int As32[2][TM * TK / 2];  // 2 x 8 KB
  __shared__ unsigned int Bs32[2][TN * TK / 2];  // 2 x 4 KB

  const int tid  = threadIdx.x;
  const int lane = tid & 31;
  const int wid  = tid >> 5;
  const int wrow = (wid >> 1) * 32;   // 0..96
  const int wcol = (wid & 1) * 32;    // 0..32
  const int m0 = blockIdx.y * TM;
  const int n0 = blockIdx.x * TN;

  v8f acc[2][2] = {};

  stage_tiles<ANCHW>(As32[0], Bs32[0], A, Bw, M, N, Kd, Sa, m0, n0, 0, tid);
  __syncthreads();

  int cur = 0;
  for (int k0 = 0; k0 < Kd; k0 += TK) {
    if (k0 + TK < Kd) {
      __builtin_prefetch(Bw + (size_t)n0 * Kd + (k0 + 2 * TK), 0, 0);
      stage_tiles<ANCHW>(As32[cur ^ 1], Bs32[cur ^ 1], A, Bw, M, N, Kd, Sa,
                         m0, n0, k0 + TK, tid);
    }

    const unsigned short* Asu = (const unsigned short*)As32[cur];
    const unsigned short* Bsu = (const unsigned short*)Bs32[cur];
    v16bf a0 = frag_ld_a(Asu, wrow,      lane);
    v16bf a1 = frag_ld_a(Asu, wrow + 16, lane);
    v16bf b0 = frag_ld_b(Bsu, wcol,      lane);
    v16bf b1 = frag_ld_b(Bsu, wcol + 16, lane);

    acc[0][0] = __builtin_amdgcn_wmma_f32_16x16x32_bf16(false, a0, false, b0, (short)0, acc[0][0], false, false);
    acc[0][1] = __builtin_amdgcn_wmma_f32_16x16x32_bf16(false, a0, false, b1, (short)0, acc[0][1], false, false);
    acc[1][0] = __builtin_amdgcn_wmma_f32_16x16x32_bf16(false, a1, false, b0, (short)0, acc[1][0], false, false);
    acc[1][1] = __builtin_amdgcn_wmma_f32_16x16x32_bf16(false, a1, false, b1, (short)0, acc[1][1], false, false);

    __syncthreads();
    cur ^= 1;
  }

  // Epilogue. C layout: vgpr r -> M = r + 8*(lane>=16), N = lane&15.
  const int moff = ((lane >> 4) & 1) * 8;
  const int nl   = lane & 15;
  const bool fullTile = ((m0 + TM) <= M) && ((n0 + TN) <= N);

  if (fullTile) {
#pragma unroll
    for (int i = 0; i < 2; ++i)
#pragma unroll
      for (int j = 0; j < 2; ++j)
#pragma unroll
        for (int r = 0; r < 8; ++r) {
          int gm = m0 + wrow + i * 16 + r + moff;
          int gn = n0 + wcol + j * 16 + nl;
          store_elem<ONCHW>(C, bias, resid, relu, N, So, gm, gn, acc[i][j][r]);
        }
  } else {
#pragma unroll
    for (int i = 0; i < 2; ++i)
#pragma unroll
      for (int j = 0; j < 2; ++j)
#pragma unroll
        for (int r = 0; r < 8; ++r) {
          int gm = m0 + wrow + i * 16 + r + moff;
          int gn = n0 + wcol + j * 16 + nl;
          if (gm < M && gn < N)
            store_elem<ONCHW>(C, bias, resid, relu, N, So, gm, gn, acc[i][j][r]);
        }
  }
}

// ---------------------------------------------------------------------------
// Support kernels (elementwise / gather / scan)
// ---------------------------------------------------------------------------
#define GRID_STRIDE(i, total)                                                 \
  for (size_t i = (size_t)blockIdx.x * blockDim.x + threadIdx.x; i < (total); \
       i += (size_t)gridDim.x * blockDim.x)

__global__ void im2col_3x3_s2(const float* __restrict__ in, float* __restrict__ out,
                              int Bn, int Cch, int Hin, int Win, int Hout, int Wout)
{
  const size_t total = (size_t)Bn * Hout * Wout * Cch * 9;
  GRID_STRIDE(i, total) {
    int col = (int)(i % (Cch * 9));
    size_t row = i / (Cch * 9);
    int kx = col % 3, ky = (col / 3) % 3, ci = col / 9;
    int ox = (int)(row % Wout);
    int oy = (int)((row / Wout) % Hout);
    int bb = (int)(row / ((size_t)Wout * Hout));
    int iy = oy * 2 - 1 + ky, ix = ox * 2 - 1 + kx;
    float v = 0.f;
    if (iy >= 0 && iy < Hin && ix >= 0 && ix < Win)
      v = in[(((size_t)bb * Cch + ci) * Hin + iy) * Win + ix];
    out[i] = v;
  }
}

// (B,D,t,f) -> rows (b*t, l) x cols (d*K+k) : seq for SRU
__global__ void unfold_kernel(const float* __restrict__ in, float* __restrict__ out,
                              int Bn, int Dch, int Tt, int Ff, int L, int Kw)
{
  const size_t total = (size_t)Bn * Tt * L * Dch * Kw;
  GRID_STRIDE(i, total) {
    int k = (int)(i % Kw);  size_t r = i / Kw;
    int d = (int)(r % Dch); r /= Dch;
    int l = (int)(r % L);   r /= L;
    int t = (int)(r % Tt);
    int b = (int)(r / Tt);
    out[i] = in[(((size_t)b * Dch + d) * Tt + t) * Ff + (l + k)];
  }
}

// SRU elementwise recurrence; writes into concatenated (rows, 2H) at col dir*H
__global__ void sru_scan(const float* __restrict__ u, const float* __restrict__ xhw,
                         const float* __restrict__ bf, const float* __restrict__ br,
                         float* __restrict__ hcat, int Nseq, int L, int Hdim, int dir)
{
  int i = blockIdx.x * blockDim.x + threadIdx.x;
  if (i >= Nseq * Hdim) return;
  int n = i / Hdim, j = i - n * Hdim;
  const float bfv = bf[j], brv = br[j];
  float c = 0.f;
  for (int s = 0; s < L; ++s) {
    int l = dir ? (L - 1 - s) : s;
    size_t ro = (size_t)n * L + l;
    const float* up = u + ro * (3 * Hdim);
    float xt = up[j];
    float f  = 1.f / (1.f + __expf(-(up[Hdim + j]     + bfv)));
    float rr = 1.f / (1.f + __expf(-(up[2 * Hdim + j] + brv)));
    float xh = xhw[ro * Hdim + j];
    c = f * c + (1.f - f) * xt;
    float h = rr * c + (1.f - rr) * xh;
    hcat[ro * (2 * Hdim) + dir * Hdim + j] = h;
  }
}

// ConvTranspose (1,K) stride 1: z laid out (B*t, I, L) -> out NCHW (B,D,t,L+K-1)
__global__ void convT_1xK(const float* __restrict__ z, const float* __restrict__ w,
                          const float* __restrict__ bias, const float* __restrict__ resid,
                          float* __restrict__ out,
                          int Bn, int Tt, int L, int Ich, int Och, int Kw)
{
  const int Fo = L + Kw - 1;
  const size_t total = (size_t)Bn * Och * Tt * Fo;
  GRID_STRIDE(i, total) {
    int ff = (int)(i % Fo); size_t r = i / Fo;
    int tt = (int)(r % Tt); r /= Tt;
    int o  = (int)(r % Och);
    int b  = (int)(r / Och);
    float acc = bias ? bias[o] : 0.f;
    const float* zrow = z + ((size_t)b * Tt + tt) * (size_t)Ich * L;
    for (int k = 0; k < Kw; ++k) {
      int l = ff - k;
      if (l < 0 || l >= L) continue;
      const float* wr = w + (size_t)o * Ich * Kw + k;
      const float* zp = zrow + l;
      for (int ci = 0; ci < Ich; ++ci)
        acc += zp[(size_t)ci * L] * wr[(size_t)ci * Kw];
    }
    if (resid) acc += resid[i];
    out[i] = acc;
  }
}

// out[bc][y][x] = in[bc][x][y] (+ resid at out index)
__global__ void transpose_hw_add(const float* __restrict__ in, const float* __restrict__ resid,
                                 float* __restrict__ out, int BC, int Yy, int Xx)
{
  const size_t total = (size_t)BC * Yy * Xx;
  GRID_STRIDE(i, total) {
    int x = (int)(i % Xx); size_t r = i / Xx;
    int y = (int)(r % Yy);
    int bc = (int)(r / Yy);
    float v = in[((size_t)bc * Xx + x) * Yy + y];
    if (resid) v += resid[i];
    out[i] = v;
  }
}

// Batched 2D transpose: out[z][c][r] = in[z][r][c]
__global__ void transpose2d(const float* __restrict__ in, float* __restrict__ out,
                            int R, int Cn, int batch)
{
  const size_t total = (size_t)batch * R * Cn;
  GRID_STRIDE(i, total) {
    int c = (int)(i % Cn); size_t t = i / Cn;
    int r = (int)(t % R);
    int z = (int)(t / R);
    out[((size_t)z * Cn + c) * R + r] = in[i];
  }
}

__global__ void split_heads(const float* __restrict__ qkv, float* __restrict__ out,
                            int Bn, int Nn, int heads, int dh, int threeD, int off)
{
  const size_t total = (size_t)Bn * heads * Nn * dh;
  GRID_STRIDE(i, total) {
    int d = (int)(i % dh); size_t r = i / dh;
    int n = (int)(r % Nn); r /= Nn;
    int h = (int)(r % heads);
    int b = (int)(r / heads);
    out[i] = qkv[((size_t)b * Nn + n) * threeD + off + h * dh + d];
  }
}

__global__ void merge_heads(const float* __restrict__ oh, float* __restrict__ out,
                            int Bn, int Nn, int heads, int dh)
{
  const size_t total = (size_t)Bn * Nn * heads * dh;
  GRID_STRIDE(i, total) {
    int d = (int)(i % dh); size_t r = i / dh;
    int h = (int)(r % heads); r /= heads;
    int n = (int)(r % Nn);
    int b = (int)(r / Nn);
    out[i] = oh[(((size_t)b * heads + h) * Nn + n) * dh + d];
  }
}

__global__ void __launch_bounds__(256)
softmax_rows(float* __restrict__ s, int rows, int cols, float scale)
{
  __shared__ float red[256];
  int row = blockIdx.x;
  if (row >= rows) return;
  float* p = s + (size_t)row * cols;
  float mx = -3.4e38f;
  for (int c = threadIdx.x; c < cols; c += 256) mx = fmaxf(mx, p[c] * scale);
  red[threadIdx.x] = mx; __syncthreads();
  for (int o = 128; o > 0; o >>= 1) {
    if ((int)threadIdx.x < o) red[threadIdx.x] = fmaxf(red[threadIdx.x], red[threadIdx.x + o]);
    __syncthreads();
  }
  mx = red[0]; __syncthreads();
  float sum = 0.f;
  for (int c = threadIdx.x; c < cols; c += 256) {
    float e = __expf(p[c] * scale - mx);
    p[c] = e; sum += e;
  }
  red[threadIdx.x] = sum; __syncthreads();
  for (int o = 128; o > 0; o >>= 1) {
    if ((int)threadIdx.x < o) red[threadIdx.x] += red[threadIdx.x + o];
    __syncthreads();
  }
  float inv = 1.f / red[0];
  for (int c = threadIdx.x; c < cols; c += 256) p[c] *= inv;
}

// ConvTranspose 2x2 stride 2 (exactly one tap per output) + residual
__global__ void convT_2x2_s2(const float* __restrict__ in, const float* __restrict__ w,
                             const float* __restrict__ bias, const float* __restrict__ resid,
                             float* __restrict__ out, int Bn, int Cch, int Hin, int Win)
{
  const int Ho = Hin * 2, Wo = Win * 2;
  const size_t total = (size_t)Bn * Cch * Ho * Wo;
  GRID_STRIDE(i, total) {
    int x = (int)(i % Wo); size_t r = i / Wo;
    int y = (int)(r % Ho); r /= Ho;
    int o = (int)(r % Cch);
    int b = (int)(r / Cch);
    int iy = y >> 1, ix = x >> 1, ky = y & 1, kx = x & 1;
    float acc = bias[o];
    const float* ip = in + ((size_t)b * Cch * Hin + iy) * Win + ix;
    const float* wp = w + (size_t)o * Cch * 4 + ky * 2 + kx;
    for (int ci = 0; ci < Cch; ++ci)
      acc += ip[(size_t)ci * Hin * Win] * wp[(size_t)ci * 4];
    if (resid) acc += resid[i];
    out[i] = acc;
  }
}

// ---------------------------------------------------------------------------
// Host orchestration
// ---------------------------------------------------------------------------
static dim3 grid1d(size_t total) {
  size_t b = (total + 255) / 256;
  if (b == 0) b = 1;
  if (b > (1u << 20)) b = (1u << 20);
  return dim3((unsigned)b);
}

extern "C" void kernel_launch(void* const* d_in, const int* in_sizes, int n_in,
                              void* d_out, int out_size, void* d_ws, size_t ws_size,
                              hipStream_t stream) {
  (void)in_sizes; (void)n_in; (void)out_size; (void)ws_size;

  const float* Ain        = (const float*)d_in[0];
  const float* compress_w = (const float*)d_in[1];
  const float* compress_b = (const float*)d_in[2];
  const float* down_w0    = (const float*)d_in[3];
  const float* down_b0    = (const float*)d_in[4];
  const float* down_w1    = (const float*)d_in[5];
  const float* down_b1    = (const float*)d_in[6];
  const float* f_Wx       = (const float*)d_in[7];
  const float* f_bf       = (const float*)d_in[8];
  const float* f_br       = (const float*)d_in[9];
  const float* f_Whw      = (const float*)d_in[10];
  const float* f_proj_w   = (const float*)d_in[11];
  const float* f_proj_b   = (const float*)d_in[12];
  const float* f_ct_w     = (const float*)d_in[13];
  const float* f_ct_b     = (const float*)d_in[14];
  const float* t_Wx       = (const float*)d_in[15];
  const float* t_bf       = (const float*)d_in[16];
  const float* t_br       = (const float*)d_in[17];
  const float* t_Whw      = (const float*)d_in[18];
  const float* t_proj_w   = (const float*)d_in[19];
  const float* t_proj_b   = (const float*)d_in[20];
  const float* t_ct_w     = (const float*)d_in[21];
  const float* t_ct_b     = (const float*)d_in[22];
  const float* qkv_w      = (const float*)d_in[23];
  const float* qkv_b      = (const float*)d_in[24];
  const float* out_w      = (const float*)d_in[25];
  const float* out_b      = (const float*)d_in[26];
  const float* up_w0      = (const float*)d_in[27];
  const float* up_b0      = (const float*)d_in[28];
  const float* up_w1      = (const float*)d_in[29];
  const float* up_b1      = (const float*)d_in[30];
  const float* final_w    = (const float*)d_in[31];
  const float* final_b    = (const float*)d_in[32];
  float* outp = (float*)d_out;

  // Bump allocator over workspace
  char* wsb = (char*)d_ws;
  size_t off = 0;
  auto alloc = [&](size_t nfloats) -> float* {
    float* p = (float*)(wsb + off);
    off += ((nfloats * sizeof(float)) + 255) & ~(size_t)255;
    return p;
  };

  auto gemm = [&](const float* Ap, const float* Bp, float* Cp, const float* biasp,
                  const float* residp, int M, int N, int Kd, int aS, int oS, int relu) {
    dim3 g((N + TN - 1) / TN, (M + TM - 1) / TM);
    if (aS > 0) {
      if (oS > 0) gemm_nt_wmma<1,1><<<g, 256, 0, stream>>>(Ap, Bp, Cp, biasp, residp, M, N, Kd, aS, oS, relu);
      else        gemm_nt_wmma<1,0><<<g, 256, 0, stream>>>(Ap, Bp, Cp, biasp, residp, M, N, Kd, aS, oS, relu);
    } else {
      if (oS > 0) gemm_nt_wmma<0,1><<<g, 256, 0, stream>>>(Ap, Bp, Cp, biasp, residp, M, N, Kd, aS, oS, relu);
      else        gemm_nt_wmma<0,0><<<g, 256, 0, stream>>>(Ap, Bp, Cp, biasp, residp, M, N, Kd, aS, oS, relu);
    }
  };

  // Derived dims
  const int S0 = TT * FFQ;                 // 32768
  const int M0 = BB * S0;                  // 262144
  const int H1 = 128, W1 = 64, S1 = H1 * W1, M1 = BB * S1;   // down0
  const int H2 = 64,  W2 = 32, S2 = H2 * W2, M2 = BB * S2;   // down1 / ag
  const int tF = 64, fF = 32, L1 = fF - KK + 1;              // freq: L=25
  const int NseqF = BB * tF, RowsF = NseqF * L1;             // 12800
  const int tT = 32, fT = 64, L2 = fT - KK + 1;              // time: L=57
  const int NseqT = BB * tT, RowsT = NseqT * L2;             // 14592
  const int Nn = tF * fF;                                    // 2048 tokens
  const int dh = DD / NHEAD;                                 // 16
  const int BH = BB * NHEAD;                                 // 32

  // --- weight pre-transposes (to (N,Kd) row-major for the NT GEMM) ---
  float* WxT_f  = alloc((size_t)2 * H3 * II);
  float* WhwT_f = alloc((size_t)2 * HH * II);
  float* projT_f= alloc((size_t)II * 2 * HH);
  float* WxT_t  = alloc((size_t)2 * H3 * II);
  float* WhwT_t = alloc((size_t)2 * HH * II);
  float* projT_t= alloc((size_t)II * 2 * HH);
  float* qkvT   = alloc((size_t)3 * DD * DD);
  float* outT   = alloc((size_t)DD * DD);
  transpose2d<<<grid1d((size_t)2*II*H3), 256, 0, stream>>>(f_Wx,  WxT_f,  II, H3, 2);
  transpose2d<<<grid1d((size_t)2*II*HH), 256, 0, stream>>>(f_Whw, WhwT_f, II, HH, 2);
  transpose2d<<<grid1d((size_t)2*HH*II), 256, 0, stream>>>(f_proj_w, projT_f, 2*HH, II, 1);
  transpose2d<<<grid1d((size_t)2*II*H3), 256, 0, stream>>>(t_Wx,  WxT_t,  II, H3, 2);
  transpose2d<<<grid1d((size_t)2*II*HH), 256, 0, stream>>>(t_Whw, WhwT_t, II, HH, 2);
  transpose2d<<<grid1d((size_t)2*HH*II), 256, 0, stream>>>(t_proj_w, projT_t, 2*HH, II, 1);
  transpose2d<<<grid1d((size_t)DD*3*DD), 256, 0, stream>>>(qkv_w, qkvT, DD, 3*DD, 1);
  transpose2d<<<grid1d((size_t)DD*DD),   256, 0, stream>>>(out_w, outT, DD, DD, 1);

  // --- activations ---
  float* x0n  = alloc((size_t)BB * DD * S0);   // (B,D,256,128)
  float* col0 = alloc((size_t)M1 * DD * 9);
  float* x1n  = alloc((size_t)BB * DD * S1);   // (B,D,128,64)
  float* col1 = alloc((size_t)M2 * DD * 9);
  float* agb  = alloc((size_t)BB * DD * S2);   // (B,D,64,32)

  // Phase 1: compress 1x1 conv (NCHW gather in, NCHW scatter out) -- WMMA GEMM
  gemm(Ain, compress_w, x0n, compress_b, nullptr, M0, DD, CC, S0, S0, 0);

  // Phase 2/3: strided 3x3 convs via im2col + WMMA GEMM (+ReLU)
  im2col_3x3_s2<<<grid1d((size_t)M1*DD*9), 256, 0, stream>>>(x0n, col0, BB, DD, TT, FFQ, H1, W1);
  gemm(col0, down_w0, x1n, down_b0, nullptr, M1, DD, DD*9, 0, S1, 1);
  im2col_3x3_s2<<<grid1d((size_t)M2*DD*9), 256, 0, stream>>>(x1n, col1, BB, DD, H1, W1, H2, W2);
  gemm(col1, down_w1, agb, down_b1, nullptr, M2, DD, DD*9, 0, S2, 1);

  // Phase 4: frequency pathway
  float* seqF  = alloc((size_t)RowsF * II);
  float* uf0   = alloc((size_t)RowsF * H3);
  float* uf1   = alloc((size_t)RowsF * H3);
  float* xhf0  = alloc((size_t)RowsF * HH);
  float* xhf1  = alloc((size_t)RowsF * HH);
  float* hcatF = alloc((size_t)RowsF * 2 * HH);
  float* zF    = alloc((size_t)RowsF * II);    // laid out (B*t, I, L)
  float* freqO = alloc((size_t)BB * DD * S2);

  unfold_kernel<<<grid1d((size_t)RowsF*II), 256, 0, stream>>>(agb, seqF, BB, DD, tF, fF, L1, KK);
  gemm(seqF, WxT_f,                   uf0, nullptr, nullptr, RowsF, H3, II, 0, 0, 0);
  gemm(seqF, WxT_f + (size_t)H3*II,   uf1, nullptr, nullptr, RowsF, H3, II, 0, 0, 0);
  gemm(seqF, WhwT_f,                  xhf0, nullptr, nullptr, RowsF, HH, II, 0, 0, 0);
  gemm(seqF, WhwT_f + (size_t)HH*II,  xhf1, nullptr, nullptr, RowsF, HH, II, 0, 0, 0);
  sru_scan<<<grid1d((size_t)NseqF*HH), 256, 0, stream>>>(uf0, xhf0, f_bf,      f_br,      hcatF, NseqF, L1, HH, 0);
  sru_scan<<<grid1d((size_t)NseqF*HH), 256, 0, stream>>>(uf1, xhf1, f_bf + HH, f_br + HH, hcatF, NseqF, L1, HH, 1);
  gemm(hcatF, projT_f, zF, f_proj_b, nullptr, RowsF, II, 2*HH, 0, L1, 0);  // -> (B*t, I, L)
  convT_1xK<<<grid1d((size_t)BB*DD*tF*fF), 256, 0, stream>>>(zF, f_ct_w, f_ct_b, agb, freqO, BB, tF, L1, II, DD, KK);

  // Phase 5: time pathway (transposed spatial)
  float* agt   = alloc((size_t)BB * DD * S2);  // (B,D,32,64)
  float* seqT  = alloc((size_t)RowsT * II);
  float* ut0   = alloc((size_t)RowsT * H3);
  float* ut1   = alloc((size_t)RowsT * H3);
  float* xht0  = alloc((size_t)RowsT * HH);
  float* xht1  = alloc((size_t)RowsT * HH);
  float* hcatT = alloc((size_t)RowsT * 2 * HH);
  float* zT    = alloc((size_t)RowsT * II);
  float* tpO   = alloc((size_t)BB * DD * S2);  // (B,D,32,64)
  float* timeO = alloc((size_t)BB * DD * S2);  // (B,D,64,32)

  transpose_hw_add<<<grid1d((size_t)BB*DD*S2), 256, 0, stream>>>(agb, nullptr, agt, BB*DD, tT, fT);
  unfold_kernel<<<grid1d((size_t)RowsT*II), 256, 0, stream>>>(agt, seqT, BB, DD, tT, fT, L2, KK);
  gemm(seqT, WxT_t,                   ut0, nullptr, nullptr, RowsT, H3, II, 0, 0, 0);
  gemm(seqT, WxT_t + (size_t)H3*II,   ut1, nullptr, nullptr, RowsT, H3, II, 0, 0, 0);
  gemm(seqT, WhwT_t,                  xht0, nullptr, nullptr, RowsT, HH, II, 0, 0, 0);
  gemm(seqT, WhwT_t + (size_t)HH*II,  xht1, nullptr, nullptr, RowsT, HH, II, 0, 0, 0);
  sru_scan<<<grid1d((size_t)NseqT*HH), 256, 0, stream>>>(ut0, xht0, t_bf,      t_br,      hcatT, NseqT, L2, HH, 0);
  sru_scan<<<grid1d((size_t)NseqT*HH), 256, 0, stream>>>(ut1, xht1, t_bf + HH, t_br + HH, hcatT, NseqT, L2, HH, 1);
  gemm(hcatT, projT_t, zT, t_proj_b, nullptr, RowsT, II, 2*HH, 0, L2, 0);
  convT_1xK<<<grid1d((size_t)BB*DD*tT*fT), 256, 0, stream>>>(zT, t_ct_w, t_ct_b, nullptr, tpO, BB, tT, L2, II, DD, KK);
  transpose_hw_add<<<grid1d((size_t)BB*DD*S2), 256, 0, stream>>>(tpO, freqO, timeO, BB*DD, tF, fF);

  // Phase 6: TF self-attention (per-(b,head) WMMA GEMMs)
  float* qkvTok = alloc((size_t)BB * Nn * 3 * DD);
  float* Qh     = alloc((size_t)BH * Nn * dh);
  float* Kh     = alloc((size_t)BH * Nn * dh);
  float* Vh     = alloc((size_t)BH * Nn * dh);
  float* VhT    = alloc((size_t)BH * dh * Nn);
  float* scores = alloc((size_t)Nn * Nn);
  float* Ohead  = alloc((size_t)BH * Nn * dh);
  float* Otok   = alloc((size_t)BB * Nn * DD);
  float* dp     = alloc((size_t)BB * DD * S2);

  gemm(timeO, qkvT, qkvTok, qkv_b, nullptr, BB * Nn, 3 * DD, DD, Nn, 0, 0);
  split_heads<<<grid1d((size_t)BH*Nn*dh), 256, 0, stream>>>(qkvTok, Qh, BB, Nn, NHEAD, dh, 3*DD, 0);
  split_heads<<<grid1d((size_t)BH*Nn*dh), 256, 0, stream>>>(qkvTok, Kh, BB, Nn, NHEAD, dh, 3*DD, DD);
  split_heads<<<grid1d((size_t)BH*Nn*dh), 256, 0, stream>>>(qkvTok, Vh, BB, Nn, NHEAD, dh, 3*DD, 2*DD);
  transpose2d<<<grid1d((size_t)BH*Nn*dh), 256, 0, stream>>>(Vh, VhT, Nn, dh, BH);

  const float scale = 1.f / sqrtf((float)dh);
  for (int z = 0; z < BH; ++z) {
    const float* Qz  = Qh  + (size_t)z * Nn * dh;
    const float* Kz  = Kh  + (size_t)z * Nn * dh;
    const float* VzT = VhT + (size_t)z * dh * Nn;
    float* Oz = Ohead + (size_t)z * Nn * dh;
    gemm(Qz, Kz, scores, nullptr, nullptr, Nn, Nn, dh, 0, 0, 0);       // Q·K^T (K=16, zero-padded)
    softmax_rows<<<Nn, 256, 0, stream>>>(scores, Nn, Nn, scale);
    gemm(scores, VzT, Oz, nullptr, nullptr, Nn, dh, Nn, 0, 0, 0);      // A·V
  }
  merge_heads<<<grid1d((size_t)BB*Nn*DD), 256, 0, stream>>>(Ohead, Otok, BB, Nn, NHEAD, dh);
  gemm(Otok, outT, dp, out_b, timeO, BB * Nn, DD, DD, 0, Nn, 0);       // + residual

  // Phase 7: U-Net reconstruction with skips
  float* up0n = alloc((size_t)BB * DD * S1);
  float* up1n = alloc((size_t)BB * DD * S0);
  convT_2x2_s2<<<grid1d((size_t)BB*DD*S1), 256, 0, stream>>>(dp,   up_w0, up_b0, x1n, up0n, BB, DD, H2, W2);
  convT_2x2_s2<<<grid1d((size_t)BB*DD*S0), 256, 0, stream>>>(up0n, up_w1, up_b1, x0n, up1n, BB, DD, H1, W1);

  // Phase 8: final 1x1 conv D->C, scattered straight into d_out (NCHW)
  gemm(up1n, final_w, outp, final_b, nullptr, M0, CC, DD, S0, S0, 0);
}
